// BTT_38431367364800
// MI455X (gfx1250) — matrix operand order repaired and checked by reading, not compiled, not run on hardware
//
#include <hip/hip_runtime.h>

// ---------------------------------------------------------------------------
// BTT layer for MI455X (gfx1250): out = x.reshape(8192,4096) @ G + bias
//   Kernel 1: build G row-by-row from the three cores; store SPLIT bf16
//             hi/lo planes, TRANSPOSED (N-major, K-contig) in workspace.
//   Kernel 2: bf16x3 WMMA GEMM (v_wmma_f32_16x16x32_bf16), fp32 accumulate,
//             software-pipelined; B tiles via async global->LDS (ASYNCcnt).
// ---------------------------------------------------------------------------

typedef __bf16 v16bf __attribute__((ext_vector_type(16)));
typedef float  v8f   __attribute__((ext_vector_type(8)));
typedef int    int4v __attribute__((ext_vector_type(4)));

#define BTT_SIZE 4096
#define BTT_M    8192

#define BM 128
#define BN 128
#define BK 32
#define PK 40   // padded ushort row stride (80 B = 16B-multiple, bank-tiling)

#if defined(__has_builtin)
#if __has_builtin(__builtin_amdgcn_global_load_async_to_lds_b128)
#define HAVE_ASYNC_LDS 1
#endif
#endif
#ifndef HAVE_ASYNC_LDS
#define HAVE_ASYNC_LDS 0
#endif

// exact parameter types of the async builtin: (V4i*1, V4i*3, Ii, Ii)
typedef __attribute__((address_space(1))) int4v* as1_q_t;
typedef __attribute__((address_space(3))) int4v* as3_q_t;

__device__ __forceinline__ void async_copy_b128(const void* gsrc, void* ldst) {
#if HAVE_ASYNC_LDS
    __builtin_amdgcn_global_load_async_to_lds_b128(
        (as1_q_t)(unsigned long long)gsrc,
        (as3_q_t)(unsigned long long)ldst, 0, 0);
#else
    (void)gsrc; (void)ldst;
#endif
}

__device__ __forceinline__ unsigned short f32_to_bf16_bits(float f) {
    union { __bf16 b; unsigned short u; } c;
    c.b = (__bf16)f;               // RNE convert
    return c.u;
}
__device__ __forceinline__ float bf16_bits_to_f32(unsigned short u) {
    union { unsigned int i; float f; } c;
    c.i = ((unsigned int)u) << 16;
    return c.f;
}
// split a,b into bf16 hi/lo; pack pairs for vector LDS stores
__device__ __forceinline__ void split2(float a, float b,
                                       unsigned int& hw, unsigned int& lw) {
    unsigned short ha = f32_to_bf16_bits(a);
    unsigned short hb = f32_to_bf16_bits(b);
    unsigned short la = f32_to_bf16_bits(a - bf16_bits_to_f32(ha));
    unsigned short lb = f32_to_bf16_bits(b - bf16_bits_to_f32(hb));
    hw = (unsigned int)ha | ((unsigned int)hb << 16);
    lw = (unsigned int)la | ((unsigned int)lb << 16);
}

__device__ __forceinline__ void wait_async_zero() {
#if HAVE_ASYNC_LDS
#if defined(__has_builtin) && __has_builtin(__builtin_amdgcn_s_wait_asynccnt)
    __builtin_amdgcn_s_wait_asynccnt(0);
#else
    asm volatile("s_wait_asynccnt 0" ::: "memory");
#endif
#endif
}

// ---------------------------------------------------------------------------
// Kernel 1: one block per G row r (4096 blocks x 128 threads).
// G[r, y*256+x*16+xp] = sum_c core2[r*128+xp*8+c] *
//                       sum_b core1[r*1024+x*64+c*8+b] * core0[r*128+y*8+b]
// Stores ghiT[n*4096+r], gloT[n*4096+r] (transposed split planes).
// ---------------------------------------------------------------------------
__global__ __launch_bounds__(128) void btt_build_g(
    const float* __restrict__ core0, const float* __restrict__ core1,
    const float* __restrict__ core2,
    unsigned short* __restrict__ ghiT, unsigned short* __restrict__ gloT) {
    const int r = blockIdx.x;
    const int t = threadIdx.x;

    __shared__ float sA0[128];    // [y][b]
    __shared__ float sA1[1024];   // [x][c][b]
    __shared__ float sA2[128];    // [xp][c]
    __shared__ float sS[2048];    // [x][c][y]

    sA0[t] = core0[(size_t)r * 128 + t];
    sA2[t] = core2[(size_t)r * 128 + t];
#pragma unroll
    for (int q = 0; q < 8; ++q)
        sA1[t + 128 * q] = core1[(size_t)r * 1024 + t + 128 * q];
    __syncthreads();

#pragma unroll
    for (int q = 0; q < 16; ++q) {
        int s = t * 16 + q;                    // s = x*128 + c*16 + y
        int x = s >> 7, c = (s >> 4) & 7, y = s & 15;
        float acc = 0.f;
#pragma unroll
        for (int b = 0; b < 8; ++b)
            acc += sA1[x * 64 + c * 8 + b] * sA0[y * 8 + b];
        sS[s] = acc;
    }
    __syncthreads();

#pragma unroll
    for (int q = 0; q < 32; ++q) {
        int n = t + 128 * q;
        int y = n >> 8, x = (n >> 4) & 15, xp = n & 15;
        float acc = 0.f;
#pragma unroll
        for (int c = 0; c < 8; ++c)
            acc += sA2[xp * 8 + c] * sS[x * 128 + c * 16 + y];
        unsigned short hi = f32_to_bf16_bits(acc);
        unsigned short lo = f32_to_bf16_bits(acc - bf16_bits_to_f32(hi));
        ghiT[(size_t)n * BTT_SIZE + r] = hi;
        gloT[(size_t)n * BTT_SIZE + r] = lo;
    }
}

// ---------------------------------------------------------------------------
// Kernel 2: bf16x3 GEMM.  grid = (N/128, M/128), 256 threads = 8 waves.
// Wave (wm,wn) owns a 64x32 output tile = 4x2 WMMA 16x16 tiles.
// A tile: single-stage LDS, register-staged + split on the fly.
// B tile: two-stage LDS, async global->LDS copies (or reg-staged fallback).
// ---------------------------------------------------------------------------
__global__ __launch_bounds__(256) void btt_gemm(
    const float* __restrict__ X,
    const unsigned short* __restrict__ ghiT,
    const unsigned short* __restrict__ gloT,
    const float* __restrict__ bias, float* __restrict__ Out) {

    __shared__ unsigned short AhiS[BM * PK];        // 10 KB
    __shared__ unsigned short AloS[BM * PK];        // 10 KB
    __shared__ unsigned short BhiS[2][BN * PK];     // 20 KB
    __shared__ unsigned short BloS[2][BN * PK];     // 20 KB

    const int tid  = threadIdx.x;
    const int lane = tid & 31;
    const int wave = tid >> 5;
    const int wm   = wave >> 2;      // 0..1
    const int wn   = wave & 3;       // 0..3
    const int half = lane >> 4;
    const int l16  = lane & 15;

    const int gm0 = blockIdx.y * BM;
    const int gn0 = blockIdx.x * BN;

    // A staging: thread -> (row, 16-wide k-half), coalesced 64 B
    const int am  = tid >> 1;
    const int akq = (tid & 1) * 16;
    // B staging: thread -> (n row pair, 8-wide k-chunk), 4 x b128 per thread
    const int bn8 = tid >> 2;        // 0..63  (also handles bn8+64)
    const int bkc = (tid & 3) * 8;   // ushort offset within row

    v8f acc[4][2];
#pragma unroll
    for (int i = 0; i < 4; ++i)
#pragma unroll
        for (int j = 0; j < 2; ++j)
            acc[i][j] = (v8f){0.f, 0.f, 0.f, 0.f, 0.f, 0.f, 0.f, 0.f};

    float areg[16];
#if !HAVE_ASYNC_LDS
    uint4 breg0, breg1, breg2, breg3;
#endif

    auto loadA = [&](int k0) {
        const float* xp = X + (size_t)(gm0 + am) * BTT_SIZE + k0 + akq;
#pragma unroll
        for (int i = 0; i < 4; ++i) {
            float4 f = ((const float4*)xp)[i];
            areg[4 * i + 0] = f.x; areg[4 * i + 1] = f.y;
            areg[4 * i + 2] = f.z; areg[4 * i + 3] = f.w;
        }
    };
    auto storeA = [&]() {
        unsigned int hw[8], lw[8];
#pragma unroll
        for (int i = 0; i < 8; ++i)
            split2(areg[2 * i], areg[2 * i + 1], hw[i], lw[i]);
        uint4* dh = (uint4*)&AhiS[am * PK + akq];
        uint4* dl = (uint4*)&AloS[am * PK + akq];
        uint4 q;
        q.x = hw[0]; q.y = hw[1]; q.z = hw[2]; q.w = hw[3]; dh[0] = q;
        q.x = hw[4]; q.y = hw[5]; q.z = hw[6]; q.w = hw[7]; dh[1] = q;
        q.x = lw[0]; q.y = lw[1]; q.z = lw[2]; q.w = lw[3]; dl[0] = q;
        q.x = lw[4]; q.y = lw[5]; q.z = lw[6]; q.w = lw[7]; dl[1] = q;
    };
    auto issueB = [&](int k0, int stg) {
        const unsigned short* g0 = ghiT + (size_t)(gn0 + bn8)      * BTT_SIZE + k0 + bkc;
        const unsigned short* g1 = ghiT + (size_t)(gn0 + 64 + bn8) * BTT_SIZE + k0 + bkc;
        const unsigned short* g2 = gloT + (size_t)(gn0 + bn8)      * BTT_SIZE + k0 + bkc;
        const unsigned short* g3 = gloT + (size_t)(gn0 + 64 + bn8) * BTT_SIZE + k0 + bkc;
#if HAVE_ASYNC_LDS
        async_copy_b128(g0, &BhiS[stg][bn8 * PK + bkc]);
        async_copy_b128(g1, &BhiS[stg][(64 + bn8) * PK + bkc]);
        async_copy_b128(g2, &BloS[stg][bn8 * PK + bkc]);
        async_copy_b128(g3, &BloS[stg][(64 + bn8) * PK + bkc]);
#else
        (void)stg;
        breg0 = *(const uint4*)g0;
        breg1 = *(const uint4*)g1;
        breg2 = *(const uint4*)g2;
        breg3 = *(const uint4*)g3;
#endif
    };
#if !HAVE_ASYNC_LDS
    auto storeB = [&](int stg) {
        *(uint4*)&BhiS[stg][bn8 * PK + bkc]        = breg0;
        *(uint4*)&BhiS[stg][(64 + bn8) * PK + bkc] = breg1;
        *(uint4*)&BloS[stg][bn8 * PK + bkc]        = breg2;
        *(uint4*)&BloS[stg][(64 + bn8) * PK + bkc] = breg3;
    };
#endif

    // ---- prologue: fill stage 0 ----
    loadA(0);
    issueB(0, 0);
    storeA();
#if !HAVE_ASYNC_LDS
    storeB(0);
#endif
    wait_async_zero();
    __syncthreads();

    // ---- main pipelined loop ----
    for (int k0 = 0, stg = 0; k0 < BTT_SIZE; k0 += BK, stg ^= 1) {
        const bool more = (k0 + BK) < BTT_SIZE;
        if (more) {
            issueB(k0 + BK, stg ^ 1);   // async into other B stage
            loadA(k0 + BK);             // global loads in flight over compute
        }

        // ---- fragments (ISA wave32 layouts) + 24 WMMA ----
        union AU { uint4 q[2]; v16bf v; };
        v16bf afh[4], afl[4];
#pragma unroll
        for (int sm = 0; sm < 4; ++sm) {
            int mb = (wm * 64 + sm * 16 + l16) * PK + 8 * half;
            AU uh, ul;
            uh.q[0] = *(const uint4*)&AhiS[mb];
            uh.q[1] = *(const uint4*)&AhiS[mb + 16];
            ul.q[0] = *(const uint4*)&AloS[mb];
            ul.q[1] = *(const uint4*)&AloS[mb + 16];
            afh[sm] = uh.v;
            afl[sm] = ul.v;
        }
#pragma unroll
        for (int sn = 0; sn < 2; ++sn) {
            int nb = (wn * 32 + sn * 16 + l16) * PK + 8 * half;
            AU bh, bl;
            bh.q[0] = *(const uint4*)&BhiS[stg][nb];
            bh.q[1] = *(const uint4*)&BhiS[stg][nb + 16];
            bl.q[0] = *(const uint4*)&BloS[stg][nb];
            bl.q[1] = *(const uint4*)&BloS[stg][nb + 16];
#pragma unroll
            for (int sm = 0; sm < 4; ++sm) {
                acc[sm][sn] = __builtin_amdgcn_wmma_f32_16x16x32_bf16(
                    false, afh[sm], false, bh.v, (short)0, acc[sm][sn],
                    false, false);
                acc[sm][sn] = __builtin_amdgcn_wmma_f32_16x16x32_bf16(
                    false, afh[sm], false, bl.v, (short)0, acc[sm][sn],
                    false, false);
                acc[sm][sn] = __builtin_amdgcn_wmma_f32_16x16x32_bf16(
                    false, afl[sm], false, bh.v, (short)0, acc[sm][sn],
                    false, false);
            }
        }

        if (more) {
            __syncthreads();            // everyone done reading A stage
            storeA();                   // convert + store next A tile
#if !HAVE_ASYNC_LDS
            storeB(stg ^ 1);
#endif
            wait_async_zero();          // B async writes landed
            __syncthreads();            // next stage visible to all
        }
    }

    // ---- epilogue: C/D layout lane=col, VGPR v -> row v + 8*half ----
#pragma unroll
    for (int sn = 0; sn < 2; ++sn) {
        int col = gn0 + wn * 32 + sn * 16 + l16;
        float bv = bias[col];
#pragma unroll
        for (int sm = 0; sm < 4; ++sm) {
            int rbase = gm0 + wm * 64 + sm * 16 + half * 8;
#pragma unroll
            for (int v = 0; v < 8; ++v)
                Out[(size_t)(rbase + v) * BTT_SIZE + col] = acc[sm][sn][v] + bv;
        }
    }
}

// ---------------------------------------------------------------------------
extern "C" void kernel_launch(void* const* d_in, const int* in_sizes, int n_in,
                              void* d_out, int out_size, void* d_ws,
                              size_t ws_size, hipStream_t stream) {
    (void)in_sizes; (void)n_in; (void)out_size; (void)ws_size;
    const float* x     = (const float*)d_in[0];
    const float* core0 = (const float*)d_in[1];
    const float* core1 = (const float*)d_in[2];
    const float* core2 = (const float*)d_in[3];
    const float* bias  = (const float*)d_in[4];

    unsigned short* ghiT = (unsigned short*)d_ws;                    // 32 MB
    unsigned short* gloT = ghiT + (size_t)BTT_SIZE * BTT_SIZE;       // 32 MB

    btt_build_g<<<BTT_SIZE, 128, 0, stream>>>(core0, core1, core2, ghiT, gloT);

    dim3 grid(BTT_SIZE / BN, BTT_M / BM);
    btt_gemm<<<grid, 256, 0, stream>>>(x, ghiT, gloT, bias, (float*)d_out);
}
